// DeltaBase_35588099014902
// MI455X (gfx1250) — matrix operand order repaired and checked by reading, not compiled, not run on hardware
//
#include <hip/hip_runtime.h>
#include <hip/hip_bf16.h>

typedef __bf16 bf16;
typedef __attribute__((ext_vector_type(8)))  bf16  v8bf;
typedef __attribute__((ext_vector_type(16))) bf16  v16bf;
typedef __attribute__((ext_vector_type(8)))  float v8f;
typedef unsigned int u32;
typedef unsigned long long u64;
typedef __attribute__((ext_vector_type(4))) u32 su32x4;
typedef __attribute__((ext_vector_type(8))) u32 su32x8;

#define BB 64
#define LL 2048
#define HH 128
#define VV 32000

// ---------------------------------------------------------------------------
// Fragment loaders for V_WMMA_F32_16X16X32_BF16 (wave32).
// A (16x32, MxK) row-major, lda=K: lane l -> m=row0+(l&15), half=l>>4,
//   element i: K = k0 + half*8 + (i%8) + (i/8)*16  => two 16B loads.
// B (32x16, KxN) stored pre-transposed Bt[N][K]: lane l -> n=col0+(l&15),
//   element i: K = k0 + half*8 + (i%8) + (i/8)*16  => two 16B loads.
// ---------------------------------------------------------------------------
__device__ __forceinline__ v16bf load_frag_a(const bf16* A, int lda, int row0,
                                             int k0, int lane) {
  int m = row0 + (lane & 15);
  int half = lane >> 4;
  const bf16* p = A + (size_t)m * lda + k0 + half * 8;
  v8bf lo = *(const v8bf*)p;
  v8bf hi = *(const v8bf*)(p + 16);
  v16bf r;
#pragma unroll
  for (int i = 0; i < 8; ++i) { r[i] = lo[i]; r[i + 8] = hi[i]; }
  return r;
}

__device__ __forceinline__ v16bf load_frag_b(const bf16* Bt, int ldb, int col0,
                                             int k0, int lane) {
  int n = col0 + (lane & 15);
  int half = lane >> 4;
  const bf16* p = Bt + (size_t)n * ldb + k0 + half * 8;
  v8bf lo = *(const v8bf*)p;
  v8bf hi = *(const v8bf*)(p + 16);
  v16bf r;
#pragma unroll
  for (int i = 0; i < 8; ++i) { r[i] = lo[i]; r[i + 8] = hi[i]; }
  return r;
}

// ---------------------------------------------------------------------------
// bf16 WMMA GEMM: C[M,N] = A[M,K] * Bt[N,K]^T (+bias)(+addsrc)(relu?)
// 256 threads = 8 waves; each wave owns a 32x64 strip (2 m-tiles x 4 n-tiles,
// 8 accumulators) so every B fragment feeds two WMMAs. M%32==0, N%64==0,
// K%32==0 at all call sites.
// ---------------------------------------------------------------------------
__global__ void wmma_gemm_kernel(const bf16* __restrict__ A,
                                 const bf16* __restrict__ Bt,
                                 const float* __restrict__ bias,
                                 const float* __restrict__ addsrc,
                                 float* __restrict__ Cf,
                                 bf16* __restrict__ Cb,
                                 int M, int N, int K, int relu) {
  int lane = threadIdx.x & 31;
  int wave = blockIdx.x * 8 + (threadIdx.x >> 5);
  int nGroups = N >> 6;
  int total = (M >> 5) * nGroups;
  if (wave >= total) return;              // wave-uniform: EXEC stays all-ones
  int row0 = (wave / nGroups) << 5;
  int col0 = (wave % nGroups) << 6;

  v8f acc[2][4] = {};
  for (int k0 = 0; k0 < K; k0 += 32) {
    v16bf a0 = load_frag_a(A, K, row0, k0, lane);
    v16bf a1 = load_frag_a(A, K, row0 + 16, k0, lane);
#pragma unroll
    for (int g = 0; g < 4; ++g) {
      v16bf b = load_frag_b(Bt, K, col0 + g * 16, k0, lane);
      acc[0][g] = __builtin_amdgcn_wmma_f32_16x16x32_bf16(
          false, a0, false, b, (short)0, acc[0][g], false, false);
      acc[1][g] = __builtin_amdgcn_wmma_f32_16x16x32_bf16(
          false, a1, false, b, (short)0, acc[1][g], false, false);
    }
  }

  // C/D layout: lane l, VGPR r -> M = row + (l>>4)*8 + r, N = col0+g*16+(l&15)
  int half = lane >> 4;
  int nl = lane & 15;
#pragma unroll
  for (int mi = 0; mi < 2; ++mi) {
#pragma unroll
    for (int g = 0; g < 4; ++g) {
      int n = col0 + g * 16 + nl;
      float bv = bias ? bias[n] : 0.0f;
#pragma unroll
      for (int r = 0; r < 8; ++r) {
        int m = row0 + mi * 16 + half * 8 + r;
        float v = acc[mi][g][r] + bv;
        if (addsrc) v += addsrc[(size_t)m * N + n];
        if (relu) v = v > 0.0f ? v : 0.0f;
        if (Cf) Cf[(size_t)m * N + n] = v;
        else    Cb[(size_t)m * N + n] = (bf16)v;
      }
    }
  }
}

// fp32 weight [K][N] -> bf16 transposed [N][K]
__global__ void convT_kernel(const float* __restrict__ W, bf16* __restrict__ Wt,
                             int K, int N) {
  int total = K * N;
  for (int idx = blockIdx.x * blockDim.x + threadIdx.x; idx < total;
       idx += gridDim.x * blockDim.x) {
    int k = idx / N, n = idx % N;
    Wt[(size_t)n * K + k] = (bf16)W[idx];
  }
}

// h = embed[seq]: fp32 copy (residual) + bf16 copy (GEMM A operand)
__global__ void gather_kernel(const int* __restrict__ seq,
                              const float* __restrict__ embed,
                              float* __restrict__ h, bf16* __restrict__ hb,
                              int total) {
  for (int idx = blockIdx.x * blockDim.x + threadIdx.x; idx < total;
       idx += gridDim.x * blockDim.x) {
    int row = idx >> 7;          // H = 128
    int col = idx & 127;
    int tok = seq[row];
    float v = embed[(size_t)tok * HH + col];
    h[idx] = v;
    hb[idx] = (bf16)v;
  }
}

// Row-wise LayerNorm over H=128, bf16 output. One 128-thread block per row.
__global__ void ln_kernel(const float* __restrict__ x,
                          const float* __restrict__ g,
                          const float* __restrict__ b,
                          bf16* __restrict__ out) {
  __shared__ float red[128];
  int row = blockIdx.x;
  int tid = threadIdx.x;
  float v = x[(size_t)row * HH + tid];
  red[tid] = v;
  __syncthreads();
#pragma unroll
  for (int s = 64; s > 0; s >>= 1) {
    if (tid < s) red[tid] += red[tid + s];
    __syncthreads();
  }
  float mu = red[0] * (1.0f / HH);
  __syncthreads();
  float d = v - mu;
  red[tid] = d * d;
  __syncthreads();
#pragma unroll
  for (int s = 64; s > 0; s >>= 1) {
    if (tid < s) red[tid] += red[tid + s];
    __syncthreads();
  }
  float var = red[0] * (1.0f / HH);
  float y = d * rsqrtf(var + 1e-5f) * g[tid] + b[tid];
  out[(size_t)row * HH + tid] = (bf16)y;
}

// ---------------------------------------------------------------------------
// Tensor Data Mover: 2D tile load (16 rows x 128 f32) global -> LDS.
// D# group0: count=1 | lds_addr | global_addr[56:0] | type=2.
// D# group1: data_size=4B, tensor_dim0=128, tensor_dim1=16, tile 128x16,
//            tensor_dim0_stride=128 (row-major k_all).
// Issued per-wave (EXEC ignored); tracked with TENSORcnt.
// Low 32 bits of a generic LDS pointer are the raw LDS byte offset
// (shared aperture lives entirely in ADDR[63:32]).
// ---------------------------------------------------------------------------
__device__ __forceinline__ void tdm_load_tile16x128(const float* gsrc,
                                                    u32 lds_off) {
  u64 ga = (u64)(uintptr_t)gsrc;
  su32x4 g0;
  g0[0] = 1u;                                   // count=1 (valid user D#)
  g0[1] = lds_off;                              // lds_addr (bytes)
  g0[2] = (u32)ga;                              // global_addr[31:0]
  g0[3] = (u32)(ga >> 32) | 0x80000000u;        // global_addr[56:32] | type=2
  su32x8 g1;
  g1[0] = 2u << 16;                             // data_size = 4 bytes
  g1[1] = 128u << 16;                           // tensor_dim0 = 128
  g1[2] = 16u << 16;                            // tensor_dim1 = 16
  g1[3] = 128u << 16;                           // tile_dim0 = 128
  g1[4] = 16u;                                  // tile_dim1 = 16, tile_dim2=0
  g1[5] = 128u;                                 // tensor_dim0_stride = 128
  g1[6] = 0u;
  g1[7] = 0u;
  asm volatile("tensor_load_to_lds %0, %1" :: "s"(g0), "s"(g1) : "memory");
}

// ---------------------------------------------------------------------------
// Sequential delta-rule scan. One block (256 thr, 8 waves) per batch.
// M[128][128] in registers: thread t=(a*16+b) owns M[a+16*ii][b+16*jj].
// k rows are streamed by the TDM in 16-row chunks, double-buffered in LDS:
// wave 0 issues tensor_load_to_lds and waits on TENSORcnt; two workgroup
// barriers per 16 steps (vs 2/step with plain loads).
// ---------------------------------------------------------------------------
__global__ void scan_kernel(const float* __restrict__ kall,
                            float* __restrict__ readout) {
  constexpr int CH = 16;                 // rows per TDM chunk
  constexpr int NCH = LL / CH;           // 128 chunks
  __shared__ float kbuf[2][CH][HH];      // 16 KB double buffer

  int batch = blockIdx.x;
  int t = threadIdx.x;
  int a = t >> 4;                        // row group 0..15
  int bc = t & 15;                       // col group 0..15
  int lane = t & 31;
  bool issuer = (t < 32);                // wave 0 (uniform per wave)

  float Mreg[8][8] = {{0.0f}};
  const float* kb = kall + (size_t)batch * LL * HH;

  if (issuer) {
    tdm_load_tile16x128(kb,            (u32)(uintptr_t)&kbuf[0][0][0]);
    tdm_load_tile16x128(kb + CH * HH,  (u32)(uintptr_t)&kbuf[1][0][0]);
  }

  for (int c = 0; c < NCH; ++c) {
    if (issuer) {
      if (c < NCH - 1) __builtin_amdgcn_s_wait_tensorcnt(1);
      else             __builtin_amdgcn_s_wait_tensorcnt(0);
    }
    __syncthreads();                     // publish chunk c to all waves

    int smax = (c == NCH - 1) ? CH - 1 : CH;   // last row of tensor is q
    for (int s = 0; s < smax; ++s) {
      const float* kv = &kbuf[c & 1][s][0];

      // wave-local ||k||^2 (every wave computes the same value)
      float ss = 0.0f;
#pragma unroll
      for (int e = 0; e < 4; ++e) { float kk = kv[lane + 32 * e]; ss += kk * kk; }
#pragma unroll
      for (int m = 16; m >= 1; m >>= 1) ss += __shfl_xor(ss, m, 32);
      float inv = 1.0f / fmaxf(sqrtf(ss), 1e-12f);

      float knj[8], ki[8];
#pragma unroll
      for (int jj = 0; jj < 8; ++jj) knj[jj] = kv[bc + 16 * jj] * inv;
#pragma unroll
      for (int ii = 0; ii < 8; ++ii) ki[ii] = kv[a + 16 * ii];

#pragma unroll
      for (int ii = 0; ii < 8; ++ii) {
        float p = 0.0f;
#pragma unroll
        for (int jj = 0; jj < 8; ++jj) p += Mreg[ii][jj] * knj[jj];
        // reduce over the 16 lanes sharing 'a' (xor masks stay in half-wave)
        p += __shfl_xor(p, 1, 32);
        p += __shfl_xor(p, 2, 32);
        p += __shfl_xor(p, 4, 32);
        p += __shfl_xor(p, 8, 32);
        float d = ki[ii] - p;            // (k - M@kn)
#pragma unroll
        for (int jj = 0; jj < 8; ++jj) Mreg[ii][jj] += d * knj[jj];
      }
    }
    __syncthreads();                     // done reading buf[c&1]
    if (issuer && (c + 2) < NCH)
      tdm_load_tile16x128(kb + (size_t)(c + 2) * CH * HH,
                          (u32)(uintptr_t)&kbuf[c & 1][0][0]);
  }

  // read = M @ q; q = last row, still resident in kbuf[(NCH-1)&1][CH-1]
  const float* qv = &kbuf[(NCH - 1) & 1][CH - 1][0];
  float qj[8];
#pragma unroll
  for (int jj = 0; jj < 8; ++jj) qj[jj] = qv[bc + 16 * jj];
#pragma unroll
  for (int ii = 0; ii < 8; ++ii) {
    float p = 0.0f;
#pragma unroll
    for (int jj = 0; jj < 8; ++jj) p += Mreg[ii][jj] * qj[jj];
    p += __shfl_xor(p, 1, 32);
    p += __shfl_xor(p, 2, 32);
    p += __shfl_xor(p, 4, 32);
    p += __shfl_xor(p, 8, 32);
    if (bc == 0) readout[batch * HH + a + 16 * ii] = p;
  }
}

// r2 = read @ rp_w + rp_b, stored bf16 (tiny 64x128 GEMM, fp32 VALU)
__global__ void small_gemm_kernel(const float* __restrict__ rd,
                                  const float* __restrict__ rp_w,
                                  const float* __restrict__ rp_b,
                                  bf16* __restrict__ r2) {
  int idx = blockIdx.x * blockDim.x + threadIdx.x;
  if (idx >= BB * HH) return;
  int b = idx >> 7;
  int n = idx & 127;
  float s = rp_b[n];
  for (int k = 0; k < HH; ++k) s += rd[b * HH + k] * rp_w[k * HH + n];
  r2[idx] = (bf16)s;
}

extern "C" void kernel_launch(void* const* d_in, const int* in_sizes, int n_in,
                              void* d_out, int out_size, void* d_ws, size_t ws_size,
                              hipStream_t stream) {
  const int*   seq   = (const int*)d_in[0];
  const float* embed = (const float*)d_in[1];
  const float* ff_w1 = (const float*)d_in[2];
  const float* ff_b1 = (const float*)d_in[3];
  const float* ff_w2 = (const float*)d_in[4];
  const float* ff_b2 = (const float*)d_in[5];
  const float* ln_g  = (const float*)d_in[6];
  const float* ln_b  = (const float*)d_in[7];
  const float* kp_w  = (const float*)d_in[8];
  const float* rp_w  = (const float*)d_in[9];
  const float* rp_b  = (const float*)d_in[10];
  const float* out_w = (const float*)d_in[11];
  const float* out_b = (const float*)d_in[12];
  float* out = (float*)d_out;

  const size_t BL = (size_t)BB * LL;
  char* w = (char*)d_ws;
  size_t off = 0;
  auto alloc = [&](size_t bytes) {
    void* p = w + off;
    off += (bytes + 255) & ~(size_t)255;
    return p;
  };
  float* h_f = (float*)alloc(BL * HH * 4);        // later reused as k_all
  bf16*  hb  = (bf16*) alloc(BL * HH * 2);        // later reused as hn
  bf16*  a1  = (bf16*) alloc(BL * 2 * HH * 2);
  float* xf  = (float*)alloc(BL * HH * 4);
  bf16*  w1t = (bf16*) alloc((size_t)2 * HH * HH * 2);
  bf16*  w2t = (bf16*) alloc((size_t)2 * HH * HH * 2);
  bf16*  kpt = (bf16*) alloc((size_t)HH * HH * 2);
  bf16*  owt = (bf16*) alloc((size_t)VV * HH * 2);
  float* rd  = (float*)alloc((size_t)BB * HH * 4);
  bf16*  r2  = (bf16*) alloc((size_t)BB * HH * 2);
  float* kall = h_f;   // reuse (h dead after FF2)
  bf16*  hn   = hb;    // reuse (hb dead after FF1)

  // weight convert + transpose to bf16 [N][K]
  convT_kernel<<<64, 256, 0, stream>>>(ff_w1, w1t, HH, 2 * HH);
  convT_kernel<<<64, 256, 0, stream>>>(ff_w2, w2t, 2 * HH, HH);
  convT_kernel<<<32, 256, 0, stream>>>(kp_w, kpt, HH, HH);
  convT_kernel<<<8192, 256, 0, stream>>>(out_w, owt, HH, VV);

  gather_kernel<<<16384, 256, 0, stream>>>(seq, embed, h_f, hb, (int)(BL * HH));

  // FF1: relu(h @ w1 + b1) -> a1 (bf16)
  {
    int waves = (int)(BL / 32) * ((2 * HH) / 64);
    wmma_gemm_kernel<<<(waves + 7) / 8, 256, 0, stream>>>(
        hb, w1t, ff_b1, nullptr, nullptr, a1, (int)BL, 2 * HH, HH, 1);
  }
  // FF2: a1 @ w2 + b2 + h -> x (fp32)
  {
    int waves = (int)(BL / 32) * (HH / 64);
    wmma_gemm_kernel<<<(waves + 7) / 8, 256, 0, stream>>>(
        a1, w2t, ff_b2, h_f, xf, nullptr, (int)BL, HH, 2 * HH, 0);
  }
  // LayerNorm -> hn (bf16)
  ln_kernel<<<(int)BL, 128, 0, stream>>>(xf, ln_g, ln_b, hn);
  // k_all = hn @ kp_w (fp32)
  {
    int waves = (int)(BL / 32) * (HH / 64);
    wmma_gemm_kernel<<<(waves + 7) / 8, 256, 0, stream>>>(
        hn, kpt, nullptr, nullptr, kall, nullptr, (int)BL, HH, HH, 0);
  }
  // sequential delta-rule scan (TDM-streamed k rows); read = M @ q
  scan_kernel<<<BB, 256, 0, stream>>>(kall, rd);
  // r2 = read @ rp_w + rp_b (bf16)
  small_gemm_kernel<<<(BB * HH + 255) / 256, 256, 0, stream>>>(rd, rp_w, rp_b, r2);
  // out = r2 @ out_w + out_b (fp32, 64 x 32000)
  {
    int waves = (BB / 32) * (VV / 64);
    wmma_gemm_kernel<<<(waves + 7) / 8, 256, 0, stream>>>(
        r2, owt, out_b, nullptr, out, nullptr, BB, VV, HH, 0);
  }
}
